// LSTMBlock_50800873177365
// MI455X (gfx1250) — matrix operand order repaired and checked by reading, not compiled, not run on hardware
//
#include <hip/hip_runtime.h>
#include <math.h>

// ---------------------------------------------------------------------------
// 2-layer LSTM, B=64, T=1024, D=1024 (gate order i,f,g,o), fp32 in/out.
//
// MI455X (gfx1250, wave32) strategy:
//  * One-time parallel conversion of x and the 4 weight matrices to bf16.
//  * One persistent kernel, 64 WGs x 256 threads (8 wave32). WG i owns hidden
//    columns [16i,16i+16) and the 4 matching gate blocks (i,f,g,o are 1024
//    apart in the 4096-wide gate vector -> cell update stays WG-local).
//  * Per step+layer: gates = [x_t | h] @ [Wih ; Whh]^T, K=2048, via
//    v_wmma_f32_16x16x32_bf16 with fp32 accumulation.
//  * Whh slices for BOTH layers pinned in LDS (264 KB of 320 KB/WGP);
//    Wih streams from L2 (all weights L2-resident, 192 MB L2).
//  * ONE device-wide barrier per timestep (atomic counter + s_sleep spin);
//    h0/h1 double-buffered so the inter-step barrier is elided.
//  * Weight-fragment loads are "laundered" with an opaque zero each step so
//    LICM cannot hoist 8 KB/lane of fragments out of the t-loop and spill
//    them to scratch (observed in round 1).
// ---------------------------------------------------------------------------

#define BB 64
#define TT 1024
#define DD 1024

#define NBLK 64     // persistent workgroups (64 * 16 cols = D)
#define NTHR 256    // 8 wave32 per WG

#define LDS_WROW  1032                 // padded LDS row stride (bf16 elems)
#define LDS_WTILE (64 * LDS_WROW)      // one layer's Whh slice (64 rows)
#define GSTRIDE   68                   // gate-exchange row stride (floats)
#define SMEM_BYTES (2 * LDS_WTILE * 2 + 64 * GSTRIDE * 4)  // 281600 B
static_assert(SMEM_BYTES <= 320 * 1024, "exceeds WGP LDS");

typedef __bf16 bf16;
typedef __attribute__((ext_vector_type(16))) __bf16 bf16x16;
typedef __attribute__((ext_vector_type(8)))  __bf16 bf16x8;
typedef __attribute__((ext_vector_type(8)))  float  f32x8;

__device__ __forceinline__ bf16x8 ld8(const bf16* p) {
  return *reinterpret_cast<const bf16x8*>(p);
}
__device__ __forceinline__ bf16x16 cat16(bf16x8 lo, bf16x8 hi) {
  return __builtin_shufflevector(lo, hi, 0,1,2,3,4,5,6,7,8,9,10,11,12,13,14,15);
}
__device__ __forceinline__ float sigmoidf_(float x) {
  return 1.0f / (1.0f + __expf(-x));
}
__device__ __forceinline__ unsigned short f2bf(float f) {
  union { float f; unsigned u; } cv; cv.f = f;
  unsigned u = cv.u;
  u += 0x7FFFu + ((u >> 16) & 1u);   // round-to-nearest-even
  return (unsigned short)(u >> 16);
}

#define WMMA_BF16(A, Bm, C) \
  __builtin_amdgcn_wmma_f32_16x16x32_bf16(false, (A), false, (Bm), (short)0, (C), false, false)

// K=1024 MACC into two 16x16 fp32 accumulators (two gate tiles, shared A).
// Per-lane pointers pre-offset for contiguous k (k = 16*half + s): a += half*16, b += half*16.
// Software-pipelined: next iteration's fragments are in flight while the
// current pair of WMMAs issues.
__device__ __forceinline__ void gemm_k1024(const bf16* a, const bf16* bA,
                                           const bf16* bB,
                                           f32x8& acc0, f32x8& acc1) {
  bf16x16 af  = cat16(ld8(a),      ld8(a + 8));
  bf16x16 bfA = cat16(ld8(bA),     ld8(bA + 8));
  bf16x16 bfB = cat16(ld8(bB),     ld8(bB + 8));
#pragma unroll 4
  for (int k = 32; k < DD; k += 32) {
    bf16x16 af_n  = cat16(ld8(a + k),  ld8(a + k + 8));
    bf16x16 bfA_n = cat16(ld8(bA + k), ld8(bA + k + 8));
    bf16x16 bfB_n = cat16(ld8(bB + k), ld8(bB + k + 8));
    acc0 = WMMA_BF16(af, bfA, acc0);
    acc1 = WMMA_BF16(af, bfB, acc1);
    af = af_n; bfA = bfA_n; bfB = bfB_n;
  }
  acc0 = WMMA_BF16(af, bfA, acc0);
  acc1 = WMMA_BF16(af, bfB, acc1);
}

__device__ __forceinline__ void grid_barrier(unsigned* cnt, unsigned target) {
  __syncthreads();
  if (threadIdx.x == 0) {
    __hip_atomic_fetch_add(cnt, 1u, __ATOMIC_RELEASE, __HIP_MEMORY_SCOPE_AGENT);
    while (__hip_atomic_load(cnt, __ATOMIC_ACQUIRE, __HIP_MEMORY_SCOPE_AGENT)
           < target) {
      __builtin_amdgcn_s_sleep(2);
    }
  }
  __syncthreads();
  __threadfence();   // agent-scope acquire for all waves in the WG
}

// One LSTM layer step for this WG's 16 d-columns.
__device__ __forceinline__ void layer_step(
    const bf16* a1, const bf16* a2,              // per-lane A-row ptrs (K=1024 each)
    const bf16* bihA, const bf16* bihB,          // Wih rows (global, per-lane)
    const bf16* bhhA, const bf16* bhhB,          // Whh rows (LDS, per-lane)
    float* gx, int tid, int d0,
    float* c_state, bf16* h_out, float* dout, int t) {
  // Opaque zero: addresses below are (re)defined inside the t-loop, so the
  // compiler cannot hoist the weight-fragment loads across timesteps (which
  // caused massive scratch spilling in round 1).
  int lz = 0;
  asm volatile("" : "+s"(lz));
  bihA += lz; bihB += lz; bhhA += lz; bhhB += lz;

  const int lane = tid & 31;
  const int half = lane >> 4;
  const int l15  = lane & 15;
  const int wv   = tid >> 5;
  const int mt   = wv & 3;
  const int gA   = (wv >> 2) * 2;
  const int gB   = gA + 1;

  f32x8 acc0 = {}, acc1 = {};
  gemm_k1024(a1, bihA, bihB, acc0, acc1);   // input/lower-h contribution
  gemm_k1024(a2, bhhA, bhhB, acc0, acc1);   // recurrent contribution (LDS W)

  __syncthreads();                           // gx free (prev layer done)
  const int mrow = mt * 16 + half * 8;       // C/D layout: M = vgpr + 8*half
#pragma unroll
  for (int v = 0; v < 8; ++v) {
    gx[(mrow + v) * GSTRIDE + gA * 16 + l15] = acc0[v];
    gx[(mrow + v) * GSTRIDE + gB * 16 + l15] = acc1[v];
  }
  __syncthreads();

  // Elementwise cell update: 64 batch x 16 cols = 1024 elems, 4 per thread.
#pragma unroll
  for (int r = 0; r < 4; ++r) {
    const int idx = tid + r * NTHR;
    const int b = idx >> 4;
    const int j = idx & 15;
    const float* grow = gx + b * GSTRIDE;
    const float ig = sigmoidf_(grow[j]);
    const float fg = sigmoidf_(grow[16 + j]);
    const float gg = tanhf(grow[32 + j]);
    const float og = sigmoidf_(grow[48 + j]);
    const int d = d0 + j;
    const float cold = c_state[(b << 10) + d];
    const float cn = fg * cold + ig * gg;
    const float hn = og * tanhf(cn);
    c_state[(b << 10) + d] = cn;
    h_out[(b << 10) + d] = (bf16)hn;
    if (dout) dout[((size_t)b * TT + t) * DD + d] = hn;
  }
}

__global__ void __launch_bounds__(NTHR)
lstm2_persistent(const bf16* __restrict__ xb,
                 const bf16* __restrict__ wih0, const bf16* __restrict__ whh0,
                 const bf16* __restrict__ wih1, const bf16* __restrict__ whh1,
                 bf16* h0b, bf16* h1b, float* c0, float* c1,
                 float* dout, unsigned* cnt) {
  extern __shared__ char smem[];
  bf16*  w0l = (bf16*)smem;
  bf16*  w1l = w0l + LDS_WTILE;
  float* gx  = (float*)(w1l + LDS_WTILE);

  const int tid = threadIdx.x;
  const int d0  = blockIdx.x * 16;

  // Pin both layers' Whh slices (64 gate-rows x 1024 k, bf16) in LDS.
  for (int idx = tid; idx < 64 * 128; idx += NTHR) {
    const int r  = idx >> 7;       // local gate-row
    const int ch = idx & 127;      // 16-byte chunk
    const int rg = ((r >> 4) << 10) + d0 + (r & 15);
    const uint4 v0 = reinterpret_cast<const uint4*>(whh0 + (size_t)rg * DD)[ch];
    const uint4 v1 = reinterpret_cast<const uint4*>(whh1 + (size_t)rg * DD)[ch];
    *reinterpret_cast<uint4*>(w0l + (size_t)r * LDS_WROW + ch * 8) = v0;
    *reinterpret_cast<uint4*>(w1l + (size_t)r * LDS_WROW + ch * 8) = v1;
  }
  __syncthreads();

  const int lane = tid & 31;
  const int half = lane >> 4;
  const int l15  = lane & 15;
  const int wv   = tid >> 5;
  const int mt   = wv & 3;
  const int gA   = (wv >> 2) * 2;
  const int gB   = gA + 1;
  const int mg   = mt * 16 + l15;       // batch row for A fragments

  // Per-lane B-row pointers (constant over t). B layout: lane%16 = n,
  // lane/16 selects K half (pre-offset by half*16).
  const bf16* bih0A = wih0 + (size_t)(gA * 1024 + d0 + l15) * DD + half * 16;
  const bf16* bih0B = wih0 + (size_t)(gB * 1024 + d0 + l15) * DD + half * 16;
  const bf16* bih1A = wih1 + (size_t)(gA * 1024 + d0 + l15) * DD + half * 16;
  const bf16* bih1B = wih1 + (size_t)(gB * 1024 + d0 + l15) * DD + half * 16;
  const bf16* bhh0A = w0l + (size_t)(gA * 16 + l15) * LDS_WROW + half * 16;
  const bf16* bhh0B = w0l + (size_t)(gB * 16 + l15) * LDS_WROW + half * 16;
  const bf16* bhh1A = w1l + (size_t)(gA * 16 + l15) * LDS_WROW + half * 16;
  const bf16* bhh1B = w1l + (size_t)(gB * 16 + l15) * LDS_WROW + half * 16;

  // A layout pre-offset: lane half picks K half (half*16).
  const bf16*  xrow = xb + (size_t)mg * TT * DD + half * 16;
  const size_t hoff = (size_t)mg * DD + half * 16;
  const int    HSZ  = BB * DD;

  for (int t = 0; t < TT; ++t) {
    const int rb = t & 1, wb = rb ^ 1;   // h double-buffer parity

    // Layer 0: gates = [x_t | h0_r] @ [Wih0 ; Whh0]^T  -> h0_w, c0
    layer_step(xrow + (size_t)t * DD, h0b + rb * HSZ + hoff,
               bih0A, bih0B, bhh0A, bhh0B,
               gx, tid, d0, c0, h0b + wb * HSZ, nullptr, t);
    __threadfence();
    grid_barrier(cnt, (unsigned)NBLK * (unsigned)(t + 1));

    // Layer 1: gates = [h0_w | h1_r] @ [Wih1 ; Whh1]^T -> h1_w, c1, out.
    // (Double buffering makes the inter-step barrier unnecessary: the next
    //  step's layer-0 only writes buffers whose last readers preceded the
    //  barrier above.)
    layer_step(h0b + wb * HSZ + hoff, h1b + rb * HSZ + hoff,
               bih1A, bih1B, bhh1A, bhh1B,
               gx, tid, d0, c1, h1b + wb * HSZ, dout, t);
  }
}

// ---------------- one-time preparation kernels ----------------
__global__ void f32x2_to_bf16x2_kernel(const float2* __restrict__ src,
                                       unsigned int* __restrict__ dst, int n2) {
  const int stride = gridDim.x * blockDim.x;
  for (int i = blockIdx.x * blockDim.x + threadIdx.x; i < n2; i += stride) {
    const float2 v = src[i];
    dst[i] = (unsigned int)f2bf(v.x) | ((unsigned int)f2bf(v.y) << 16);
  }
}

__global__ void init_state_kernel(unsigned short* hbufs, float* cbufs,
                                  unsigned* cnt) {
  const int i = blockIdx.x * blockDim.x + threadIdx.x;
  if (i == 0) *cnt = 0;
  const int stride = gridDim.x * blockDim.x;
  for (int j = i; j < 4 * BB * DD; j += stride) hbufs[j] = 0;
  for (int j = i; j < 2 * BB * DD; j += stride) cbufs[j] = 0.0f;
}

// ---------------- host launch ----------------
extern "C" void kernel_launch(void* const* d_in, const int* in_sizes, int n_in,
                              void* d_out, int out_size, void* d_ws,
                              size_t ws_size, hipStream_t stream) {
  const float* x    = (const float*)d_in[0];
  const float* Wih0 = (const float*)d_in[1];
  const float* Whh0 = (const float*)d_in[2];
  const float* Wih1 = (const float*)d_in[3];
  const float* Whh1 = (const float*)d_in[4];
  float* out = (float*)d_out;

  char* ws = (char*)d_ws;
  const size_t WBYTES = (size_t)4096 * 1024 * 2;  // one bf16 weight matrix
  unsigned* cnt = (unsigned*)(ws + 0);
  unsigned short* wih0b = (unsigned short*)(ws + 256);
  unsigned short* whh0b = (unsigned short*)(ws + 256 + 1 * WBYTES);
  unsigned short* wih1b = (unsigned short*)(ws + 256 + 2 * WBYTES);
  unsigned short* whh1b = (unsigned short*)(ws + 256 + 3 * WBYTES);
  unsigned short* xb    = (unsigned short*)(ws + 256 + 4 * WBYTES);
  const size_t XBYTES = (size_t)BB * TT * DD * 2;
  unsigned short* h0b = (unsigned short*)((char*)xb + XBYTES);
  unsigned short* h1b = h0b + 2 * BB * DD;
  float* c0 = (float*)(h1b + 2 * BB * DD);
  float* c1 = c0 + BB * DD;

  (void)hipFuncSetAttribute((const void*)lstm2_persistent,
                            hipFuncAttributeMaxDynamicSharedMemorySize,
                            SMEM_BYTES);

  const int WN2 = 4096 * 1024 / 2;
  f32x2_to_bf16x2_kernel<<<1024, NTHR, 0, stream>>>(
      (const float2*)Wih0, (unsigned int*)wih0b, WN2);
  f32x2_to_bf16x2_kernel<<<1024, NTHR, 0, stream>>>(
      (const float2*)Whh0, (unsigned int*)whh0b, WN2);
  f32x2_to_bf16x2_kernel<<<1024, NTHR, 0, stream>>>(
      (const float2*)Wih1, (unsigned int*)wih1b, WN2);
  f32x2_to_bf16x2_kernel<<<1024, NTHR, 0, stream>>>(
      (const float2*)Whh1, (unsigned int*)whh1b, WN2);
  f32x2_to_bf16x2_kernel<<<4096, NTHR, 0, stream>>>(
      (const float2*)x, (unsigned int*)xb, BB * TT * DD / 2);
  init_state_kernel<<<256, NTHR, 0, stream>>>(h0b, c0, cnt);

  lstm2_persistent<<<NBLK, NTHR, SMEM_BYTES, stream>>>(
      (const bf16*)xb, (const bf16*)wih0b, (const bf16*)whh0b,
      (const bf16*)wih1b, (const bf16*)whh1b,
      (bf16*)h0b, (bf16*)h1b, c0, c1, out, cnt);
}
